// MambaBlock_50105088475400
// MI455X (gfx1250) — matrix operand order repaired and checked by reading, not compiled, not run on hardware
//
#include <hip/hip_runtime.h>

#define DIMM 1024
#define DI   2048      // d_inner
#define NB   4         // batch
#define LL   2048      // seq len
#define MROWS (NB * LL)  // 8192

#define BM 128
#define BN 64
#define BK 64
#define LDS_S 72   // LDS row stride (bf16): 144B = 9*16 (b128-aligned), 36 banks (conflict-spread)

typedef __bf16 v16bf  __attribute__((ext_vector_type(16)));
typedef float  v8f    __attribute__((ext_vector_type(8)));
typedef __bf16 bf16x4 __attribute__((ext_vector_type(4)));
typedef __bf16 bf16x8 __attribute__((ext_vector_type(8)));

__device__ __forceinline__ float siluf(float v) {
    return v / (1.0f + __expf(-v));
}

__device__ __forceinline__ float softplusf(float v) {
    return (v > 20.0f) ? v : __logf(1.0f + __expf(v));
}

// Async DMA one 16-byte chunk global -> LDS (no VGPR roundtrip, tracked by ASYNCcnt).
__device__ __forceinline__ void async_copy_b128(void* ldsp, const void* gp) {
    unsigned lds_off = (unsigned)(uintptr_t)ldsp;   // LDS byte offset (addr[31:0])
    asm volatile("global_load_async_to_lds_b128 %0, %1, off"
                 :: "v"(lds_off), "v"(gp) : "memory");
}

// C[M,N] = A[M,K] @ B[K,N]; A is bf16 row-major (M x K), Bt is bf16 B-TRANSPOSED (N x K).
// fp32 accumulate/output. act==1: C = softplus(C + bias[col]).
// Block = 128 threads (4 waves); each wave computes a 32x64 C tile (8 WMMA / 6 frag loads).
// Requires M%128==0, N%64==0, K%64==0, K/64 >= 2.
__global__ __launch_bounds__(128) void gemm_bf16_async(
    const __bf16* __restrict__ A, int lda,
    const __bf16* __restrict__ Bt, int ldb,
    float* __restrict__ C, int ldc,
    int N, int K,
    const float* __restrict__ bias, int act)
{
    __shared__ __bf16 sA[2][BM * LDS_S];   // [row][k]
    __shared__ __bf16 sB[2][BN * LDS_S];   // [col][k]

    const int tid  = threadIdx.x;
    const int wave = tid >> 5;          // 0..3: 32-row strip; all waves span the 64 cols
    const int lane = tid & 31;
    const int half = lane >> 4;
    const int l16  = lane & 15;
    const int rowBase = blockIdx.y * BM;
    const int colBase = blockIdx.x * BN;

    v8f acc[8];                         // [ai*4 + bi], 32x64 per wave
#pragma unroll
    for (int j = 0; j < 8; ++j)
#pragma unroll
        for (int e = 0; e < 8; ++e) acc[j][e] = 0.0f;

    // Staging: A tile 128x64 bf16 = 512 b128 chunks (4/thread), B tile 64x64 = 256 (2/thread).
    auto stage = [&](int buf, int k0) {
#pragma unroll
        for (int i = 0; i < 4; ++i) {
            int chunk = tid + i * 128;
            int r = chunk >> 2, c = chunk & 3;
            async_copy_b128(&sA[buf][r * LDS_S + c * 16],
                            A + (size_t)(rowBase + r) * lda + k0 + c * 16);
        }
#pragma unroll
        for (int i = 0; i < 2; ++i) {
            int chunk = tid + i * 128;
            int r = chunk >> 2, c = chunk & 3;
            async_copy_b128(&sB[buf][r * LDS_S + c * 16],
                            Bt + (size_t)(colBase + r) * ldb + k0 + c * 16);
        }
    };

    auto compute = [&](const __bf16* __restrict__ Ab, const __bf16* __restrict__ Bb) {
#pragma unroll
        for (int kk = 0; kk < BK; kk += 32) {
            // A fragments (16x32, ISA 7.12.2): row = l16, K groups by lane-half -> b128 pairs.
            v16bf a[2];
#pragma unroll
            for (int ai = 0; ai < 2; ++ai) {
                int row = wave * 32 + ai * 16 + l16;
                bf16x8 lo = *(const bf16x8*)&Ab[row * LDS_S + kk + half * 8];
                bf16x8 hi = *(const bf16x8*)&Ab[row * LDS_S + kk + 16 + half * 8];
#pragma unroll
                for (int e = 0; e < 8; ++e) { a[ai][e] = lo[e]; a[ai][8 + e] = hi[e]; }
            }
            // B fragments (32x16): col = l16, K groups by lane-half (same as A) -> b128 pairs.
            v16bf b[4];
#pragma unroll
            for (int bi = 0; bi < 4; ++bi) {
                int col = bi * 16 + l16;
                bf16x8 lo = *(const bf16x8*)&Bb[col * LDS_S + kk + half * 8];
                bf16x8 hi = *(const bf16x8*)&Bb[col * LDS_S + kk + 16 + half * 8];
#pragma unroll
                for (int e = 0; e < 8; ++e) { b[bi][e] = lo[e]; b[bi][8 + e] = hi[e]; }
            }
#pragma unroll
            for (int ai = 0; ai < 2; ++ai)
#pragma unroll
                for (int bi = 0; bi < 4; ++bi)
                    acc[ai * 4 + bi] = __builtin_amdgcn_wmma_f32_16x16x32_bf16(
                        false, a[ai], false, b[bi], (short)0, acc[ai * 4 + bi], false, false);
        }
    };

    const int nk = K / BK;
    stage(0, 0);                         // prologue: tile 0 -> buffer 0
    for (int t = 0; t < nk - 1; ++t) {
        stage((t + 1) & 1, (t + 1) * BK);
        asm volatile("s_wait_asynccnt 0x6" ::: "memory");  // tile t resident, t+1 in flight
        __syncthreads();
        compute(sA[t & 1], sB[t & 1]);
        __syncthreads();
    }
    asm volatile("s_wait_asynccnt 0x0" ::: "memory");
    __syncthreads();
    compute(sA[(nk - 1) & 1], sB[(nk - 1) & 1]);

    // C/D layout: VGPR r -> m = r + 8*half, n = l16.
#pragma unroll
    for (int ai = 0; ai < 2; ++ai) {
#pragma unroll
        for (int bi = 0; bi < 4; ++bi) {
            int col = colBase + bi * 16 + l16;
#pragma unroll
            for (int r = 0; r < 8; ++r) {
                int row = rowBase + wave * 32 + ai * 16 + half * 8 + r;
                float v = acc[ai * 4 + bi][r];
                if (act == 1) v = softplusf(v + bias[col]);
                C[(size_t)row * ldc + col] = v;
            }
        }
    }
}

// fp32 -> bf16 packed copy (count must be divisible by 1024).
__global__ __launch_bounds__(256) void cvt_bf16_kernel(
    const float* __restrict__ in, __bf16* __restrict__ out)
{
    int i4 = blockIdx.x * 256 + threadIdx.x;
    float4 v = ((const float4*)in)[i4];
    bf16x4 b;
    b[0] = (__bf16)v.x; b[1] = (__bf16)v.y; b[2] = (__bf16)v.z; b[3] = (__bf16)v.w;
    ((bf16x4*)out)[i4] = b;
}

// fp32 (K x N) -> bf16 transposed (N x K): out[n*K + k] = in[k*N + n].
__global__ __launch_bounds__(256) void cvt_bf16_T_kernel(
    const float* __restrict__ in, __bf16* __restrict__ out, int K, int N)
{
    int gid = blockIdx.x * 256 + threadIdx.x;
    int k = gid % K;
    int n = gid / K;
    out[(size_t)n * K + k] = (__bf16)in[(size_t)k * N + n];
}

// Causal depthwise conv (K=4) + bias + SiLU. Reads xc half of xz (stride 4096),
// writes packed fp32 xc (for scan/xproj) and bf16 xc (for WMMA GEMM).
__global__ __launch_bounds__(256) void conv_silu_kernel(
    const float* __restrict__ xz, const float* __restrict__ cw,
    const float* __restrict__ cb, float* __restrict__ xc,
    __bf16* __restrict__ xc_bf)
{
    int gid = blockIdx.x * 256 + threadIdx.x;   // over B*L*DI
    int d  = gid & (DI - 1);
    int bl = gid >> 11;                          // b*L + l
    int l  = bl & (LL - 1);
    float s = cb[d];
#pragma unroll
    for (int k = 0; k < 4; ++k) {
        int ls = l - 3 + k;
        float xv = (ls >= 0) ? xz[(size_t)(bl - 3 + k) * (2 * DI) + d] : 0.0f;
        s += cw[d * 4 + k] * xv;
    }
    float r = siluf(s);
    xc[(size_t)bl * DI + d] = r;
    xc_bf[(size_t)bl * DI + d] = (__bf16)r;
}

// B_ssm[row, n] = sum_d xc[row,d] * W_xproj[d, 16+n]  (n = 0..15)
__global__ __launch_bounds__(256) void xproj_kernel(
    const float* __restrict__ xc, const float* __restrict__ W,
    float* __restrict__ Bss)
{
    int gid = blockIdx.x * 256 + threadIdx.x;   // MROWS*16
    int n   = gid & 15;
    int row = gid >> 4;
    const float* xr = xc + (size_t)row * DI;
    float s = 0.0f;
    for (int d = 0; d < DI; d += 4) {
        s += xr[d + 0] * W[(size_t)(d + 0) * 32 + 16 + n];
        s += xr[d + 1] * W[(size_t)(d + 1) * 32 + 16 + n];
        s += xr[d + 2] * W[(size_t)(d + 2) * 32 + 16 + n];
        s += xr[d + 3] * W[(size_t)(d + 3) * 32 + 16 + n];
    }
    Bss[(size_t)row * 16 + n] = s;
}

// Selective scan, 4 lanes per (b,d) channel: lane-quad q owns states n = 4q..4q+3.
// States evolve independently; only y = sum_n s*B needs a width-4 shfl_xor butterfly.
// Reads z half of xz; writes bf16 y*silu(z) (A-input of the final WMMA GEMM).
__global__ __launch_bounds__(256) void scan_kernel(
    const float* __restrict__ delta, const float* __restrict__ xc,
    const float* __restrict__ Bss, const float* __restrict__ A_log,
    const float* __restrict__ Dp, const float* __restrict__ xz,
    __bf16* __restrict__ y_bf)
{
    int gid = blockIdx.x * 256 + threadIdx.x;   // 4 * B * DI = 32768 threads
    int q  = gid & 3;                            // state quad within channel
    int ch = gid >> 2;                           // channel = b*DI + d
    int d = ch & (DI - 1);
    int b = ch >> 11;

    float Ar[4];
#pragma unroll
    for (int j = 0; j < 4; ++j) Ar[j] = -__expf(A_log[d * 16 + q * 4 + j]);
    float Dv = Dp[d];

    float s[4] = {0.0f, 0.0f, 0.0f, 0.0f};

    for (int l = 0; l < LL; ++l) {
        size_t bl = (size_t)b * LL + l;
        float dt = delta[bl * DI + d];
        float xv = xc[bl * DI + d];
        float du = dt * xv;
        float4 Bq = ((const float4*)(Bss + bl * 16))[q];
        float Bj[4] = { Bq.x, Bq.y, Bq.z, Bq.w };
        float yp = 0.0f;
#pragma unroll
        for (int j = 0; j < 4; ++j) {
            s[j] = __expf(dt * Ar[j]) * s[j] + du * Bj[j];
            yp += s[j] * Bj[j];
        }
        yp += __shfl_xor(yp, 1, 4);
        yp += __shfl_xor(yp, 2, 4);
        if (q == 0) {
            float z = xz[bl * (2 * DI) + DI + d];
            y_bf[bl * DI + d] = (__bf16)((yp + Dv * xv) * siluf(z));
        }
    }
}

extern "C" void kernel_launch(void* const* d_in, const int* in_sizes, int n_in,
                              void* d_out, int out_size, void* d_ws, size_t ws_size,
                              hipStream_t stream) {
    (void)in_sizes; (void)n_in; (void)out_size; (void)ws_size;
    const float* x       = (const float*)d_in[0];
    const float* W_in    = (const float*)d_in[1];
    const float* conv_w  = (const float*)d_in[2];
    const float* conv_b  = (const float*)d_in[3];
    const float* W_xproj = (const float*)d_in[4];
    const float* W_dt    = (const float*)d_in[5];
    const float* b_dt    = (const float*)d_in[6];
    const float* A_log   = (const float*)d_in[7];
    const float* Dp      = (const float*)d_in[8];
    const float* W_out   = (const float*)d_in[9];
    float* out = (float*)d_out;

    // fp32 workspace region
    float* ws    = (float*)d_ws;
    float* xz    = ws;                                   // 8192 x 4096
    float* xc    = xz + (size_t)MROWS * (2 * DI);        // 8192 x 2048
    float* delta = xc + (size_t)MROWS * DI;              // 8192 x 2048
    float* bssm  = delta + (size_t)MROWS * DI;           // 8192 x 16
    // bf16 workspace region
    __bf16* bf     = (__bf16*)(bssm + (size_t)MROWS * 16);
    __bf16* x_bf   = bf;                                 // 8192 x 1024
    __bf16* WinT   = x_bf + (size_t)MROWS * DIMM;        // 4096 x 1024 (W_in^T)
    __bf16* WdtT   = WinT + (size_t)(2 * DI) * DIMM;     // 2048 x 2048 (W_dt^T)
    __bf16* WoutT  = WdtT + (size_t)DI * DI;             // 1024 x 2048 (W_out^T)
    __bf16* xc_bf  = WoutT + (size_t)DIMM * DI;          // 8192 x 2048
    __bf16* y_bf   = xc_bf + (size_t)MROWS * DI;         // 8192 x 2048

    // 0) one-time bf16 conversions (weights transposed so GEMM B-tiles are K-contiguous)
    cvt_bf16_kernel<<<(MROWS * DIMM) / 1024, 256, 0, stream>>>(x, x_bf);
    cvt_bf16_T_kernel<<<(DIMM * 2 * DI) / 256, 256, 0, stream>>>(W_in, WinT, DIMM, 2 * DI);
    cvt_bf16_T_kernel<<<(DI * DI) / 256, 256, 0, stream>>>(W_dt, WdtT, DI, DI);
    cvt_bf16_T_kernel<<<(DI * DIMM) / 256, 256, 0, stream>>>(W_out, WoutT, DI, DIMM);

    // 1) xz = x @ W_in   (M=8192, N=4096, K=1024)
    gemm_bf16_async<<<dim3((2 * DI) / BN, MROWS / BM), 128, 0, stream>>>(
        x_bf, DIMM, WinT, DIMM, xz, 2 * DI, 2 * DI, DIMM, nullptr, 0);

    // 2) depthwise conv + SiLU (fp32 + bf16 outputs)
    conv_silu_kernel<<<(MROWS * DI) / 256, 256, 0, stream>>>(xz, conv_w, conv_b, xc, xc_bf);

    // 3) delta = softplus(xc @ W_dt + b_dt)  (M=8192, N=2048, K=2048)
    gemm_bf16_async<<<dim3(DI / BN, MROWS / BM), 128, 0, stream>>>(
        xc_bf, DI, WdtT, DI, delta, DI, DI, DI, b_dt, 1);

    // 4) B_ssm = (xc @ W_xproj)[:, 16:32]
    xproj_kernel<<<(MROWS * 16) / 256, 256, 0, stream>>>(xc, W_xproj, bssm);

    // 5) selective scan (4 lanes per channel) -> bf16 y
    scan_kernel<<<(4 * NB * DI) / 256, 256, 0, stream>>>(delta, xc, bssm, A_log, Dp, xz, y_bf);

    // 6) out = y @ W_out  (M=8192, N=1024, K=2048)
    gemm_bf16_async<<<dim3(DIMM / BN, MROWS / BM), 128, 0, stream>>>(
        y_bf, DI, WoutT, DI, out, DIMM, DIMM, DI, nullptr, 0);
}
